// TransCorrBlock_45191645889235
// MI455X (gfx1250) — matrix operand order repaired and checked by reading, not compiled, not run on hardware
//
#include <hip/hip_runtime.h>

typedef _Float16 half_t;
typedef __attribute__((ext_vector_type(16))) _Float16 v16h;
typedef __attribute__((ext_vector_type(2)))  _Float16 h2;
typedef __attribute__((ext_vector_type(8)))  float    v8f;

namespace {
constexpr int Bv = 2, Cc = 256, Hh = 46, Ww = 96;
constexpr int HWs = Hh * Ww;          // 4416
constexpr int NI  = Bv * HWs;         // 8832
constexpr int Dd  = 256, D2 = 512;
constexpr int H1 = 23, W1 = 48, H2 = 11, W2 = 24, H3 = 5, W3 = 12;
constexpr int SCP = HWs + 4;          // padded LDS row stride (bank-conflict avoidance)
}

union Frag { v16h v; _Float16 h[16]; };

// ---- WMMA fragment loaders (layouts per CDNA5 ISA 7.12.2) --------------------
// A: 16x32 f16, row-major source (ld halves/row).
// lane<16 : M=lane, K = 0..7 (h0..7) and 16..23 (h8..15)
// lane>=16: M=lane-16, K = 8..15 and 24..31
__device__ __forceinline__ v16h load_frag_a(const half_t* base, int ld) {
  int lane = threadIdx.x & 31;
  const half_t* p = base + (long)(lane & 15) * ld + ((lane < 16) ? 0 : 8);
  Frag a;
#pragma unroll
  for (int t = 0; t < 8; ++t) { a.h[t] = p[t]; a.h[8 + t] = p[16 + t]; }
  return a.v;
}
// B operand B[k][n] = Rows[n][k] (computing X @ Rowsᵀ). lane = n (mod 16),
// lanes 0-15 hold K=0..7,16..23, lanes 16-31 hold K=8..15,24..31 (same as A).
__device__ __forceinline__ v16h load_frag_b(const half_t* rows, int ld) {
  int lane = threadIdx.x & 31;
  const half_t* p = rows + (long)(lane & 15) * ld + ((lane < 16) ? 0 : 8);
  Frag b;
#pragma unroll
  for (int t = 0; t < 8; ++t) { b.h[t] = p[t]; b.h[8 + t] = p[16 + t]; }
  return b.v;
}

// ---- fp32 -> f16 tiled transpose: dst[c][r] = src[r][c] ----------------------
__global__ void k_transpose_to_h(const float* __restrict__ src, half_t* __restrict__ dst,
                                 int rows, int cols, long sstride, long dstride) {
  __shared__ float tile[32][33];
  src += (long)blockIdx.z * sstride;
  dst += (long)blockIdx.z * dstride;
  int c0 = blockIdx.x * 32, r0 = blockIdx.y * 32;
  for (int i = threadIdx.y; i < 32; i += 8) {
    int r = r0 + i, c = c0 + threadIdx.x;
    tile[i][threadIdx.x] = (r < rows && c < cols) ? src[(long)r * cols + c] : 0.f;
  }
  __syncthreads();
  for (int i = threadIdx.y; i < 32; i += 8) {
    int c = c0 + i, r = r0 + threadIdx.x;
    if (c < cols && r < rows) dst[(long)c * rows + r] = (half_t)tile[threadIdx.x][i];
  }
}

// ---- vispos = Fᵀ@Wf + bf + pos  (WMMA f16->f32), out f16 ---------------------
// Fh: [NI][C] f16, WT: [D][C] f16 (Wfᵀ), out: [NI][D] f16
__global__ void __launch_bounds__(128)
k_vispos(const half_t* __restrict__ Fh, const half_t* __restrict__ WT,
         const float* __restrict__ bf, const float* __restrict__ Wpos,
         const float* __restrict__ coords, half_t* __restrict__ out) {
  int wave = threadIdx.x >> 5, lane = threadIdx.x & 31;
  long rowBase = (long)blockIdx.x * 16;
  int colBase = (blockIdx.y * 4 + wave) * 16;
  v8f acc = {};
#pragma unroll 4
  for (int k = 0; k < Cc; k += 32) {
    v16h a = load_frag_a(Fh + rowBase * Cc + k, Cc);
    v16h b = load_frag_b(WT + (long)colBase * Cc + k, Cc);
    acc = __builtin_amdgcn_wmma_f32_16x16x32_f16(false, a, false, b, (short)0, acc, false, false);
  }
  int n = lane & 15, mh = (lane < 16) ? 0 : 8;
  int col = colBase + n;
  float bfv = bf[col], w0 = Wpos[col], w1 = Wpos[Dd + col];
#pragma unroll
  for (int r = 0; r < 8; ++r) {
    long i = rowBase + mh + r;
    int bI = (int)(i / HWs), iL = (int)(i - (long)bI * HWs);
    float x, y;
    if (coords) { x = coords[(long)bI * 2 * HWs + iL]; y = coords[(long)(bI * 2 + 1) * HWs + iL]; }
    else        { y = (float)(iL / Ww); x = (float)(iL % Ww); }
    out[i * Dd + col] = (half_t)(acc[r] + bfv + (y * w0 + x * w1));
  }
}

// ---- q = vispos@Wq, pack Qcat = [q | pos]  (likewise k/Kcat) -----------------
__global__ void __launch_bounds__(128)
k_qk(const half_t* __restrict__ Vh, const half_t* __restrict__ WT,
     const float* __restrict__ Wpos, const float* __restrict__ coords,
     half_t* __restrict__ cat) {
  int wave = threadIdx.x >> 5, lane = threadIdx.x & 31;
  long rowBase = (long)blockIdx.x * 16;
  int colBase = (blockIdx.y * 4 + wave) * 16;
  v8f acc = {};
#pragma unroll 4
  for (int k = 0; k < Dd; k += 32) {
    v16h a = load_frag_a(Vh + rowBase * Dd + k, Dd);
    v16h b = load_frag_b(WT + (long)colBase * Dd + k, Dd);
    acc = __builtin_amdgcn_wmma_f32_16x16x32_f16(false, a, false, b, (short)0, acc, false, false);
  }
  int n = lane & 15, mh = (lane < 16) ? 0 : 8;
  int col = colBase + n;
  float w0 = Wpos[col], w1 = Wpos[Dd + col];
#pragma unroll
  for (int r = 0; r < 8; ++r) {
    long i = rowBase + mh + r;
    int bI = (int)(i / HWs), iL = (int)(i - (long)bI * HWs);
    float x, y;
    if (coords) { x = coords[(long)bI * 2 * HWs + iL]; y = coords[(long)(bI * 2 + 1) * HWs + iL]; }
    else        { y = (float)(iL / Ww); x = (float)(iL % Ww); }
    cat[i * D2 + col]      = (half_t)acc[r];
    cat[i * D2 + Dd + col] = (half_t)(y * w0 + x * w1);   // pos term (K=512 fused GEMM)
  }
}

// ---- fused scores GEMM (K=512) + row softmax, 16 rows/WG, stripe in LDS ------
// 299 KB LDS => one WG per WGP (2 waves/SIMD), so A fragments (128 VGPRs) are
// register-resident: steady state is 32 b128 loads + 16 back-to-back WMMAs/tile.
__global__ void __launch_bounds__(256)
k_scores(const half_t* __restrict__ Qc, const half_t* __restrict__ Kc,
         half_t* __restrict__ corr) {
  extern __shared__ char smem[];
  float*  sc = (float*)smem;                                       // [16][SCP]
  half_t* qt = (half_t*)(smem + (size_t)16 * SCP * sizeof(float)); // [16][512]
  int rowTile = blockIdx.x;                 // 552 tiles, never straddle a batch
  int b = rowTile / (HWs / 16);
  long rowBase = (long)rowTile * 16;
  { // stage Q tile (16x512 f16 = 16 KB)
    const float4* s = (const float4*)(Qc + rowBase * D2);
    float4* d = (float4*)qt;
    for (int t = threadIdx.x; t < 16 * D2 / 8; t += blockDim.x) d[t] = s[t];
  }
  __syncthreads();
  int wave = threadIdx.x >> 5, lane = threadIdx.x & 31;
  int n = lane & 15, mh = (lane < 16) ? 0 : 8;
  const half_t* Kb = Kc + (long)b * HWs * D2;
  // Hoist all 16 A fragments (K=512) out of the column loop: 128 VGPRs.
  v16h af[16];
#pragma unroll
  for (int ks = 0; ks < 16; ++ks) af[ks] = load_frag_a(qt + ks * 32, D2);
  for (int ct = wave; ct < HWs / 16; ct += 8) {
    long jBase = (long)ct * 16;
    if (ct + 8 < HWs / 16)   // stream next K tile for this wave (global_prefetch_b8)
      __builtin_prefetch(Kb + (jBase + 128) * D2 + (long)n * D2, 0, 1);
    v8f acc = {};
#pragma unroll
    for (int ks = 0; ks < 16; ++ks) {
      v16h bf_ = load_frag_b(Kb + jBase * D2 + ks * 32, D2);
      acc = __builtin_amdgcn_wmma_f32_16x16x32_f16(false, af[ks], false, bf_,
                                                   (short)0, acc, false, false);
    }
#pragma unroll
    for (int r = 0; r < 8; ++r)
      sc[(mh + r) * SCP + jBase + n] = acc[r] * 0.0625f;      // /sqrt(256)
  }
  __syncthreads();
  for (int row = wave * 2; row < wave * 2 + 2; ++row) {       // 8 waves x 2 rows
    float* s = sc + (long)row * SCP;
    float mx = -3.0e38f;
    for (int j = lane; j < HWs; j += 32) mx = fmaxf(mx, s[j]);
#pragma unroll
    for (int off = 16; off; off >>= 1) mx = fmaxf(mx, __shfl_xor(mx, off, 32));
    float sum = 0.f;
    for (int j = lane; j < HWs; j += 32) { float e = __expf(s[j] - mx); s[j] = e; sum += e; }
#pragma unroll
    for (int off = 16; off; off >>= 1) sum += __shfl_xor(sum, off, 32);
    float inv = 1.f / sum;
    half_t* o = corr + (rowBase + row) * (long)HWs;
    for (int j = lane * 2; j < HWs; j += 64) {                // packed b32 stores
      h2 pk = { (half_t)(s[j] * inv), (half_t)(s[j + 1] * inv) };
      *(h2*)(o + j) = pk;
    }
  }
}

// ---- 2x2 avg pool, stride 2, VALID (f16 in/out, fp32 math) -------------------
__global__ void k_pool(const half_t* __restrict__ in, half_t* __restrict__ out,
                       int Hi, int Wi, int Ho, int Wo, long total) {
  long tid = (long)blockIdx.x * blockDim.x + threadIdx.x;
  if (tid >= total) return;
  int xo = (int)(tid % Wo); long t = tid / Wo;
  int yo = (int)(t % Ho);   long img = t / Ho;
  const half_t* p = in + img * (long)(Hi * Wi) + (long)(2 * yo) * Wi + 2 * xo;
  float v = ((float)p[0] + (float)p[1] + (float)p[Wi] + (float)p[Wi + 1]) * 0.25f;
  out[img * (long)(Ho * Wo) + (long)yo * Wo + xo] = (half_t)v;
}

// ---- multiscale 9x9 bilinear lookup; one WG per pixel, pyramid slice in LDS --
__global__ void __launch_bounds__(128)
k_lookup(const half_t* __restrict__ l0, const half_t* __restrict__ l1,
         const half_t* __restrict__ l2, const half_t* __restrict__ l3,
         const float* __restrict__ cl, float* __restrict__ out) {
  __shared__ float sh[HWs + H1 * W1 + H2 * W2 + H3 * W3];     // 5844 floats
  int img = blockIdx.x;
  int b = img / HWs, iL = img - b * HWs;
  const half_t* srcs[4] = { l0 + (long)img * HWs, l1 + (long)img * (H1 * W1),
                            l2 + (long)img * (H2 * W2), l3 + (long)img * (H3 * W3) };
  const int offs[5] = { 0, HWs, HWs + H1 * W1, HWs + H1 * W1 + H2 * W2,
                        HWs + H1 * W1 + H2 * W2 + H3 * W3 };
  for (int lv = 0; lv < 4; ++lv)
    for (int t = threadIdx.x; t < offs[lv + 1] - offs[lv]; t += 128)
      sh[offs[lv] + t] = (float)srcs[lv][t];
  __syncthreads();
  float cx = cl[(long)b * 2 * HWs + iL];
  float cy = cl[(long)(b * 2 + 1) * HWs + iL];
  const int Hl[4] = { Hh, H1, H2, H3 }, Wl[4] = { Ww, W1, W2, W3 };
  for (int t = threadIdx.x; t < 324; t += 128) {
    int l = t / 81, kk = t - l * 81, ky = kk / 9, kx = kk - ky * 9;
    float inv = 1.f / (float)(1 << l);
    // reference adds delta=(dy,dx) onto (x,y): x gets d[ky], y gets d[kx]
    float xs = cx * inv + (float)(ky - 4);
    float ys = cy * inv + (float)(kx - 4);
    const float* im = sh + offs[l];
    int Wc = Wl[l], Hc = Hl[l];
    float x0 = floorf(xs), y0 = floorf(ys);
    float wx1 = xs - x0, wx0 = 1.f - wx1, wy1 = ys - y0, wy0 = 1.f - wy1;
    auto corner = [&](float xi, float yi) -> float {
      bool valid = (xi >= 0.f) && (xi <= (float)(Wc - 1)) && (yi >= 0.f) && (yi <= (float)(Hc - 1));
      int xc = (int)fminf(fmaxf(xi, 0.f), (float)(Wc - 1));
      int yc = (int)fminf(fmaxf(yi, 0.f), (float)(Hc - 1));
      return valid ? im[yc * Wc + xc] : 0.f;
    };
    float v = corner(x0, y0) * wx0 * wy0 + corner(x0 + 1.f, y0) * wx1 * wy0
            + corner(x0, y0 + 1.f) * wx0 * wy1 + corner(x0 + 1.f, y0 + 1.f) * wx1 * wy1;
    out[((long)b * 324 + t) * HWs + iL] = v;
  }
}

extern "C" void kernel_launch(void* const* d_in, const int* in_sizes, int n_in,
                              void* d_out, int out_size, void* d_ws, size_t ws_size,
                              hipStream_t stream) {
  const float* fmap1 = (const float*)d_in[0];
  const float* fmap2 = (const float*)d_in[1];
  const float* coords1 = (const float*)d_in[2];
  const float* clook  = (const float*)d_in[3];
  const float* Wf  = (const float*)d_in[4];
  const float* bf  = (const float*)d_in[5];
  const float* Wpos = (const float*)d_in[6];
  const float* Wq  = (const float*)d_in[7];
  const float* Wk  = (const float*)d_in[8];
  float* out = (float*)d_out;

  char* ws = (char*)d_ws;
  size_t off = 0;
  auto take = [&](size_t bytes) { char* p = ws + off; off += (bytes + 255) & ~(size_t)255; return p; };
  half_t* F1h = (half_t*)take((size_t)NI * Cc * 2);
  half_t* F2h = (half_t*)take((size_t)NI * Cc * 2);
  half_t* WfT = (half_t*)take((size_t)Cc * Dd * 2);
  half_t* WqT = (half_t*)take((size_t)Dd * Dd * 2);
  half_t* WkT = (half_t*)take((size_t)Dd * Dd * 2);
  half_t* V1h = (half_t*)take((size_t)NI * Dd * 2);
  half_t* V2h = (half_t*)take((size_t)NI * Dd * 2);
  half_t* Qcat = (half_t*)take((size_t)NI * D2 * 2);
  half_t* Kcat = (half_t*)take((size_t)NI * D2 * 2);
  half_t* P0 = (half_t*)take((size_t)NI * HWs * 2);
  half_t* P1 = (half_t*)take((size_t)NI * H1 * W1 * 2);
  half_t* P2 = (half_t*)take((size_t)NI * H2 * W2 * 2);
  half_t* P3 = (half_t*)take((size_t)NI * H3 * W3 * 2);

  dim3 tb(32, 8);
  // fmaps: per-batch [C][HW] -> [HW][C] f16
  k_transpose_to_h<<<dim3((HWs + 31) / 32, (Cc + 31) / 32, Bv), tb, 0, stream>>>(
      fmap1, F1h, Cc, HWs, (long)Cc * HWs, (long)HWs * Cc);
  k_transpose_to_h<<<dim3((HWs + 31) / 32, (Cc + 31) / 32, Bv), tb, 0, stream>>>(
      fmap2, F2h, Cc, HWs, (long)Cc * HWs, (long)HWs * Cc);
  // weights: [C][D] -> [D][C]
  k_transpose_to_h<<<dim3(8, 8, 1), tb, 0, stream>>>(Wf, WfT, Cc, Dd, 0, 0);
  k_transpose_to_h<<<dim3(8, 8, 1), tb, 0, stream>>>(Wq, WqT, Dd, Dd, 0, 0);
  k_transpose_to_h<<<dim3(8, 8, 1), tb, 0, stream>>>(Wk, WkT, Dd, Dd, 0, 0);

  dim3 gg(NI / 16, Dd / 64);
  k_vispos<<<gg, 128, 0, stream>>>(F1h, WfT, bf, Wpos, coords1, V1h);
  k_vispos<<<gg, 128, 0, stream>>>(F2h, WfT, bf, Wpos, nullptr, V2h);
  k_qk<<<gg, 128, 0, stream>>>(V1h, WqT, Wpos, coords1, Qcat);
  k_qk<<<gg, 128, 0, stream>>>(V2h, WkT, Wpos, nullptr, Kcat);

  size_t lds = (size_t)16 * SCP * sizeof(float) + (size_t)16 * D2 * 2; // 299,264 B (<320 KB WGP LDS)
  (void)hipFuncSetAttribute((const void*)k_scores,
                            hipFuncAttributeMaxDynamicSharedMemorySize, (int)lds);
  k_scores<<<dim3(NI / 16), 256, lds, stream>>>(Qcat, Kcat, P0);

  long t1 = (long)NI * H1 * W1, t2 = (long)NI * H2 * W2, t3 = (long)NI * H3 * W3;
  k_pool<<<dim3((unsigned)((t1 + 255) / 256)), 256, 0, stream>>>(P0, P1, Hh, Ww, H1, W1, t1);
  k_pool<<<dim3((unsigned)((t2 + 255) / 256)), 256, 0, stream>>>(P1, P2, H1, W1, H2, W2, t2);
  k_pool<<<dim3((unsigned)((t3 + 255) / 256)), 256, 0, stream>>>(P2, P3, H2, W2, H3, W3, t3);

  k_lookup<<<dim3(NI), 128, 0, stream>>>(P0, P1, P2, P3, clook, out);
}
